// Contrast_44521630990983
// MI455X (gfx1250) — compile-verified, matrix operand order checked by reading, not run on hardware
//
#include <hip/hip_runtime.h>
#include <hip/hip_bf16.h>

#define NROWS 8192
#define HDIM  512
#define TAU   0.8f
#define LAM   0.5f
#define EPSV  1e-8f

typedef unsigned short ushort_t;
typedef __attribute__((ext_vector_type(16))) __bf16 bf16x16;
typedef __attribute__((ext_vector_type(8)))  float  floatx8;

union Frag {
    bf16x16 v;
    uint4   q[2];
};

__device__ __forceinline__ ushort_t f32_to_bf16(float f) {
    unsigned u = __float_as_uint(f);
    u += 0x7FFFu + ((u >> 16) & 1u);   // round-to-nearest-even
    return (ushort_t)(u >> 16);
}

// ---------------------------------------------------------------- converts
__global__ __launch_bounds__(256) void cvt_kernel(const float* __restrict__ src,
                                                  ushort_t* __restrict__ dst, int n) {
    int i = blockIdx.x * 256 + threadIdx.x;
    if (i < n) dst[i] = f32_to_bf16(src[i]);
}

__global__ __launch_bounds__(256) void zero_kernel(float* __restrict__ p, int n) {
    int i = blockIdx.x * 256 + threadIdx.x;
    if (i < n) p[i] = 0.0f;
}

// ---------------------------------------------------------------- projection GEMM
// C[row, col] = act( sum_k A[row,k] * W[col,k] + bias[col] )
// MODE 0: ELU -> bf16 out ; MODE 1: identity -> f32 out
// Block: 128 threads = 4 waves; block tile = 16 rows x 256 cols (64 cols/wave).
template<int MODE>
__global__ __launch_bounds__(128) void gemm_kernel(const ushort_t* __restrict__ A,
                                                   const ushort_t* __restrict__ W,
                                                   const float* __restrict__ bias,
                                                   ushort_t* __restrict__ outb,
                                                   float* __restrict__ outf) {
    const int lane = threadIdx.x & 31;
    const int wave = threadIdx.x >> 5;
    const int n    = lane & 15;
    const int hi   = lane >> 4;
    const int row0 = blockIdx.x * 16;
    const int col0 = blockIdx.y * 256 + wave * 64;

    floatx8 acc[4] = {};
#pragma unroll 4
    for (int k = 0; k < HDIM; k += 32) {
        Frag a;
        const uint4* ap = (const uint4*)(A + (size_t)(row0 + n) * HDIM + k + hi * 8);
        a.q[0] = ap[0];
        a.q[1] = ap[2];
#pragma unroll
        for (int t = 0; t < 4; ++t) {
            Frag b;
            const uint4* bp = (const uint4*)(W + (size_t)(col0 + t * 16 + n) * HDIM + k + hi * 8);
            b.q[0] = bp[0];
            b.q[1] = bp[2];
            acc[t] = __builtin_amdgcn_wmma_f32_16x16x32_bf16(
                false, a.v, false, b.v, (short)0, acc[t], false, false);
        }
    }
#pragma unroll
    for (int t = 0; t < 4; ++t) {
        const int col = col0 + t * 16 + n;
        const float bv = bias[col];
#pragma unroll
        for (int v = 0; v < 8; ++v) {
            const int row = row0 + v + 8 * hi;
            float x = acc[t][v] + bv;
            if (MODE == 0) {
                float y = x > 0.0f ? x : (__expf(x) - 1.0f);   // ELU(alpha=1)
                outb[(size_t)row * HDIM + col] = f32_to_bf16(y);
            } else {
                outf[(size_t)row * HDIM + col] = x;
            }
        }
    }
}

// ---------------------------------------------------------------- row normalize
// zn[i] = zp[i] / (||zp[i]|| * sqrt(tau))  -> so dot(zn_a, zn_b) = cos/tau
__global__ __launch_bounds__(256) void normalize_kernel(const float* __restrict__ zp,
                                                        ushort_t* __restrict__ zn) {
    const int lane = threadIdx.x & 31;
    const int wave = threadIdx.x >> 5;
    const int row  = blockIdx.x * 8 + wave;
    const float* p = zp + (size_t)row * HDIM;
    float ss = 0.0f;
    for (int i = lane; i < HDIM; i += 32) { float v = p[i]; ss += v * v; }
#pragma unroll
    for (int m = 16; m; m >>= 1) ss += __shfl_xor(ss, m, 32);
    const float scale = rsqrtf(ss * TAU);
    ushort_t* q = zn + (size_t)row * HDIM;
    for (int i = lane; i < HDIM; i += 32) q[i] = f32_to_bf16(p[i] * scale);
}

// ---------------------------------------------------------------- fused sim tile
// Block tile 128x128; 8 waves, each computing a 32x64 sub-tile (2x4 WMMA frags)
// -> 8 WMMA per K-step vs 12 b128 loads (1.5 loads/WMMA).
// K loop unrolled x4 only: keeps ~48 b128 loads in flight while staying under
// 256 VGPRs (avoids s_set_vgpr_msb bank-switch churn, keeps occupancy up).
// For each element m[a][b] = exp(dot(zn_mp[a], zn_sc[b])):
//   rowsum[a] += m ; rowpos[a] += m*pos[a][b] ; colsum[b] += m ; colpos[b] += m*pos[b][a]
__global__ __launch_bounds__(256) void sim_kernel(const ushort_t* __restrict__ Zm,
                                                  const ushort_t* __restrict__ Zs,
                                                  const int* __restrict__ pos,
                                                  float* __restrict__ rowsum,
                                                  float* __restrict__ rowpos,
                                                  float* __restrict__ colsum,
                                                  float* __restrict__ colpos) {
    // pos tiles staged as bytes (values are 0/1). Row stride 132 bytes (=33 words):
    // direct reads are lane-consecutive; transposed reads hit distinct banks.
    __shared__ unsigned char posA[128 * 132];   // pos[row0+r][col0+c]
    __shared__ unsigned char posB[128 * 132];   // pos[col0+r][row0+c] (transposed region)

    const int row0 = blockIdx.y * 128;
    const int col0 = blockIdx.x * 128;

    for (int idx = threadIdx.x; idx < 128 * 128; idx += 256) {
        const int r = idx >> 7, c = idx & 127;
        posA[r * 132 + c] = (unsigned char)pos[(size_t)(row0 + r) * NROWS + (col0 + c)];
        posB[r * 132 + c] = (unsigned char)pos[(size_t)(col0 + r) * NROWS + (row0 + c)];
    }
    __syncthreads();

    const int lane = threadIdx.x & 31;
    const int wave = threadIdx.x >> 5;   // 0..7
    const int rt   = wave & 3;           // 4 row groups of 32
    const int cc   = wave >> 2;          // 2 col groups of 64
    const int n    = lane & 15;
    const int hi   = lane >> 4;

    const int rbase = rt * 32;           // local row base of this wave
    const int cbase = cc * 64;           // local col base of this wave

    floatx8 acc[2][4] = {};
#pragma unroll 4
    for (int k = 0; k < HDIM; k += 32) {
        Frag a[2], b[4];
#pragma unroll
        for (int r = 0; r < 2; ++r) {
            const uint4* ap =
                (const uint4*)(Zm + (size_t)(row0 + rbase + r * 16 + n) * HDIM + k + hi * 8);
            a[r].q[0] = ap[0];
            a[r].q[1] = ap[2];
        }
#pragma unroll
        for (int c = 0; c < 4; ++c) {
            const uint4* bp =
                (const uint4*)(Zs + (size_t)(col0 + cbase + c * 16 + n) * HDIM + k + hi * 8);
            b[c].q[0] = bp[0];
            b[c].q[1] = bp[2];
        }
#pragma unroll
        for (int r = 0; r < 2; ++r)
#pragma unroll
            for (int c = 0; c < 4; ++c)
                acc[r][c] = __builtin_amdgcn_wmma_f32_16x16x32_bf16(
                    false, a[r].v, false, b[c].v, (short)0, acc[r][c], false, false);
    }

    // ---- epilogue: exp + masked row/col reductions
    float rowS[2][8] = {}, rowP[2][8] = {};
#pragma unroll
    for (int c = 0; c < 4; ++c) {
        const int ncol_l = cbase + c * 16 + n;        // local col 0..127
        float csum = 0.0f, cposv = 0.0f;
#pragma unroll
        for (int r = 0; r < 2; ++r) {
#pragma unroll
            for (int v = 0; v < 8; ++v) {
                const int mrow_l = rbase + r * 16 + v + 8 * hi;   // local row 0..127
                const float e = __expf(acc[r][c][v]);
                csum  += e;
                cposv += e * (float)posB[ncol_l * 132 + mrow_l];
                rowS[r][v] += e;
                rowP[r][v] += e * (float)posA[mrow_l * 132 + ncol_l];
            }
        }
        csum  += __shfl_xor(csum, 16, 32);
        cposv += __shfl_xor(cposv, 16, 32);
        if (hi == 0) {
            atomicAdd(&colsum[col0 + ncol_l], csum);
            atomicAdd(&colpos[col0 + ncol_l], cposv);
        }
    }
#pragma unroll
    for (int r = 0; r < 2; ++r) {
#pragma unroll
        for (int v = 0; v < 8; ++v) {
            float rs = rowS[r][v], rp = rowP[r][v];
#pragma unroll
            for (int m = 8; m; m >>= 1) {
                rs += __shfl_xor(rs, m, 16);
                rp += __shfl_xor(rp, m, 16);
            }
            if (n == 0) {
                const int grow = row0 + rbase + r * 16 + v + 8 * hi;
                atomicAdd(&rowsum[grow], rs);
                atomicAdd(&rowpos[grow], rp);
            }
        }
    }
}

// ---------------------------------------------------------------- finalize
__global__ __launch_bounds__(256) void finalize_kernel(const float* __restrict__ rowsum,
                                                       const float* __restrict__ rowpos,
                                                       const float* __restrict__ colsum,
                                                       const float* __restrict__ colpos,
                                                       float* __restrict__ out) {
    __shared__ float sm[256], ss[256];
    float amp = 0.0f, asc = 0.0f;
    for (int i = threadIdx.x; i < NROWS; i += 256) {
        amp += logf(rowpos[i] / (rowsum[i] + EPSV));
        asc += logf(colpos[i] / (colsum[i] + EPSV));
    }
    sm[threadIdx.x] = amp; ss[threadIdx.x] = asc;
    __syncthreads();
    for (int s = 128; s; s >>= 1) {
        if ((int)threadIdx.x < s) { sm[threadIdx.x] += sm[threadIdx.x + s]; ss[threadIdx.x] += ss[threadIdx.x + s]; }
        __syncthreads();
    }
    if (threadIdx.x == 0) {
        const float lmp = -sm[0] / (float)NROWS;
        const float lsc = -ss[0] / (float)NROWS;
        out[0] = LAM * lmp + (1.0f - LAM) * lsc;
    }
}

// ---------------------------------------------------------------- launcher
extern "C" void kernel_launch(void* const* d_in, const int* in_sizes, int n_in,
                              void* d_out, int out_size, void* d_ws, size_t ws_size,
                              hipStream_t stream) {
    const float* z_mp = (const float*)d_in[0];
    const float* z_sc = (const float*)d_in[1];
    const int*   pos  = (const int*)d_in[2];
    const float* W1   = (const float*)d_in[3];
    const float* b1   = (const float*)d_in[4];
    const float* W2   = (const float*)d_in[5];
    const float* b2   = (const float*)d_in[6];
    float* out = (float*)d_out;

    char* ws = (char*)d_ws;
    // workspace layout (bytes)
    ushort_t* Wbf1   = (ushort_t*)(ws + 0x0000000);            // 512 KB
    ushort_t* Wbf2   = (ushort_t*)(ws + 0x0080000);            // 512 KB
    ushort_t* zbf_mp = (ushort_t*)(ws + 0x0100000);            // 8 MB
    ushort_t* zbf_sc = (ushort_t*)(ws + 0x0900000);            // 8 MB
    ushort_t* h_mp   = (ushort_t*)(ws + 0x1100000);            // 8 MB
    ushort_t* h_sc   = (ushort_t*)(ws + 0x1900000);            // 8 MB
    float*    zp_mp  = (float*)   (ws + 0x2100000);            // 16 MB
    float*    zp_sc  = (float*)   (ws + 0x3100000);            // 16 MB
    ushort_t* zn_mp  = (ushort_t*)(ws + 0x4100000);            // 8 MB
    ushort_t* zn_sc  = (ushort_t*)(ws + 0x4900000);            // 8 MB
    float*    acc    = (float*)   (ws + 0x5100000);            // 4 * 32 KB
    float* rowsum = acc;
    float* rowpos = acc + NROWS;
    float* colsum = acc + 2 * NROWS;
    float* colpos = acc + 3 * NROWS;

    const int nz = NROWS * HDIM;     // 4194304
    const int nw = HDIM * HDIM;      // 262144

    // 1) convert inputs to bf16
    cvt_kernel<<<(nw + 255) / 256, 256, 0, stream>>>(W1, Wbf1, nw);
    cvt_kernel<<<(nw + 255) / 256, 256, 0, stream>>>(W2, Wbf2, nw);
    cvt_kernel<<<(nz + 255) / 256, 256, 0, stream>>>(z_mp, zbf_mp, nz);
    cvt_kernel<<<(nz + 255) / 256, 256, 0, stream>>>(z_sc, zbf_sc, nz);

    // 2) projection: h = elu(z @ W1^T + b1) ; zp = h @ W2^T + b2
    dim3 gG(NROWS / 16, HDIM / 256);
    gemm_kernel<0><<<gG, 128, 0, stream>>>(zbf_mp, Wbf1, b1, h_mp, nullptr);
    gemm_kernel<0><<<gG, 128, 0, stream>>>(zbf_sc, Wbf1, b1, h_sc, nullptr);
    gemm_kernel<1><<<gG, 128, 0, stream>>>(h_mp, Wbf2, b2, nullptr, zp_mp);
    gemm_kernel<1><<<gG, 128, 0, stream>>>(h_sc, Wbf2, b2, nullptr, zp_sc);

    // 3) normalize rows (fold 1/||z|| and 1/sqrt(tau))
    normalize_kernel<<<NROWS / 8, 256, 0, stream>>>(zp_mp, zn_mp);
    normalize_kernel<<<NROWS / 8, 256, 0, stream>>>(zp_sc, zn_sc);

    // 4) zero accumulators (must be re-done every call)
    zero_kernel<<<(4 * NROWS + 255) / 256, 256, 0, stream>>>(acc, 4 * NROWS);

    // 5) fused similarity tiles + masked reductions (128x128 tiles)
    dim3 gS(NROWS / 128, NROWS / 128);
    sim_kernel<<<gS, 256, 0, stream>>>(zn_mp, zn_sc, pos, rowsum, rowpos, colsum, colpos);

    // 6) scalar loss
    finalize_kernel<<<1, 256, 0, stream>>>(rowsum, rowpos, colsum, colpos, out);

    (void)in_sizes; (void)n_in; (void)out_size; (void)ws_size;
}
